// AttentionDecoder_70360154243535
// MI455X (gfx1250) — matrix operand order repaired and checked by reading, not compiled, not run on hardware
//
#include <hip/hip_runtime.h>

typedef __bf16 v16bf __attribute__((ext_vector_type(16)));
typedef __bf16 v8bf  __attribute__((ext_vector_type(8)));
typedef float  v8f   __attribute__((ext_vector_type(8)));
typedef float  v4f   __attribute__((ext_vector_type(4)));

#define B_   64
#define T_   64
#define E_   256
#define S_   128
#define ENC_ 1024
#define RNN_ 1024
#define ATT_ 1024
#define DENSE_ 8000
#define KCAT_ 2304   // ENC + E + RNN
#define NGATE_ 4096  // 4*RNN

__device__ __forceinline__ float fast_tanh(float x) {
    // 1 - 2/(e^{2x}+1): monotone, no NaN at +/-inf
    float e = __expf(2.0f * x);
    return 1.0f - 2.0f / (e + 1.0f);
}
__device__ __forceinline__ float fast_sigmoid(float x) {
    return 1.0f / (1.0f + __expf(-x));
}

// ---------------------------------------------------------------------------
// Generic WMMA GEMM:  C[M,N](f32 or bf16) = A[M,K] * BT[N,K]^T + bias[N]
// A is bf16 row-major (lda) unless A_F32 (then fp32, converted in-flight).
// Each wave computes a 16x64 strip: one A fragment feeds 4 WMMA accumulators.
// Requires M%16==0, N%64==0, K%32==0 (true for every call site here).
// ---------------------------------------------------------------------------
template <bool A_F32, bool OUT_BF16>
__global__ void gemm_wmma_kernel(const void* __restrict__ Av,
                                 const __bf16* __restrict__ BT,
                                 const float* __restrict__ bias,
                                 void* __restrict__ Cv,
                                 int M, int N, int K, int lda, int ldc) {
    const int wid  = (blockIdx.x * blockDim.x + threadIdx.x) >> 5;
    const int lane = threadIdx.x & 31;
    const int tilesM  = M >> 4;
    const int tilesN4 = N >> 6;
    if (wid >= tilesM * tilesN4) return;   // whole-wave guard: EXEC stays all-1 for WMMA
    const int tm = wid % tilesM;
    const int tn = wid / tilesM;
    const int m0 = tm << 4;
    const int n0 = tn << 6;
    const int row  = lane & 15;   // M (A) / N (B,C) index within tile
    const int half = lane >> 4;

    const __bf16* Ab = (const __bf16*)Av;
    const float*  Af = (const float*)Av;
    const int m = m0 + row;

    v8f acc[4] = {v8f{0}, v8f{0}, v8f{0}, v8f{0}};

    for (int k0 = 0; k0 < K; k0 += 32) {
        // --- A fragment (16-bit A 16x32, contiguous k order, same as B):
        // lanes 0-15:  elems 0..15 = K k0..k0+15
        // lanes 16-31: elems 0..15 = K k0+16..k0+31
        v16bf a;
        if (A_F32) {
            const float* p0 = Af + (size_t)m * lda + k0 + 16 * half;
            const float* p1 = p0 + 8;
            v4f f0 = *(const v4f*)(p0);
            v4f f1 = *(const v4f*)(p0 + 4);
            v4f f2 = *(const v4f*)(p1);
            v4f f3 = *(const v4f*)(p1 + 4);
#pragma unroll
            for (int j = 0; j < 4; ++j) {
                a[j]      = (__bf16)f0[j];
                a[4 + j]  = (__bf16)f1[j];
                a[8 + j]  = (__bf16)f2[j];
                a[12 + j] = (__bf16)f3[j];
            }
        } else {
            const __bf16* pa = Ab + (size_t)m * lda + k0 + 16 * half;
            v8bf lo = *(const v8bf*)(pa);
            v8bf hi = *(const v8bf*)(pa + 8);
            a = __builtin_shufflevector(lo, hi, 0, 1, 2, 3, 4, 5, 6, 7,
                                        8, 9, 10, 11, 12, 13, 14, 15);
        }
        // --- B fragments: BT is [N,K] row-major; lane=column, half selects K 0-15/16-31
        const __bf16* bp = BT + (size_t)(n0 + row) * K + k0 + 16 * half;
#pragma unroll
        for (int j = 0; j < 4; ++j) {
            v16bf bj = *(const v16bf*)(bp + (size_t)(16 * j) * K);
            acc[j] = __builtin_amdgcn_wmma_f32_16x16x32_bf16(
                false, a, false, bj, (short)0, acc[j], false, false);
        }
    }

    // --- C store: VGPR r -> M = m0 + 8*half + r ; N = n0 + 16*j + (lane&15)
#pragma unroll
    for (int j = 0; j < 4; ++j) {
        const int n = n0 + 16 * j + row;
        const float bv = bias ? bias[n] : 0.0f;
#pragma unroll
        for (int r = 0; r < 8; ++r) {
            const int mm = m0 + 8 * half + r;
            const float val = acc[j][r] + bv;
            if (OUT_BF16)
                ((__bf16*)Cv)[(size_t)mm * ldc + n] = (__bf16)val;
            else
                ((float*)Cv)[(size_t)mm * ldc + n] = val;
        }
    }
}

// ---------------------------------------------------------------------------
// Bahdanau attention for one batch row per block (256 threads):
// score_s = V . tanh(q + keys[s]); softmax over S; ctx = sum_s attn_s*enc[s].
// Writes ctx (bf16) into A2[b, 0:1024] and x_t (bf16) into A2[b, 1024:1280].
// ---------------------------------------------------------------------------
__global__ void attn_ctx_kernel(const float* __restrict__ q,
                                const __bf16* __restrict__ keys,
                                const float* __restrict__ V,
                                const float* __restrict__ bV,
                                const float* __restrict__ enc,
                                const __bf16* __restrict__ xbf,
                                __bf16* __restrict__ A2, int t) {
    const int b = blockIdx.x;
    const int tid = threadIdx.x;            // 256 threads
    __shared__ float part[256];
    __shared__ float attn[S_];
    __shared__ float red[64];

    // --- scores: 2 threads per s, each covering 512 of the 1024 ATT dims
    {
        const int s = tid >> 1, h = tid & 1;
        const float*  qb = q + (size_t)b * ATT_ + h * 512;
        const __bf16* kb = keys + ((size_t)b * S_ + s) * ATT_ + h * 512;
        const float*  vb = V + h * 512;
        float sum = 0.0f;
        for (int a = 0; a < 512; ++a)
            sum += fast_tanh(qb[a] + (float)kb[a]) * vb[a];
        part[tid] = sum;
    }
    __syncthreads();
    if (tid < S_) attn[tid] = part[2 * tid] + part[2 * tid + 1] + bV[0];
    __syncthreads();

    // --- softmax over S=128
    if (tid < 64) red[tid] = fmaxf(attn[tid], attn[tid + 64]);
    __syncthreads();
    for (int off = 32; off > 0; off >>= 1) {
        if (tid < off) red[tid] = fmaxf(red[tid], red[tid + off]);
        __syncthreads();
    }
    const float mx = red[0];
    if (tid < S_) attn[tid] = __expf(attn[tid] - mx);
    __syncthreads();
    if (tid < 64) red[tid] = attn[tid] + attn[tid + 64];
    __syncthreads();
    for (int off = 32; off > 0; off >>= 1) {
        if (tid < off) red[tid] = red[tid] + red[tid + off];
        __syncthreads();
    }
    const float inv = 1.0f / red[0];
    __syncthreads();
    if (tid < S_) attn[tid] *= inv;
    __syncthreads();

    // --- ctx: each thread accumulates 4 consecutive ENC columns over S
    {
        v4f acc = {0.f, 0.f, 0.f, 0.f};
        const float* eb = enc + (size_t)b * S_ * ENC_ + tid * 4;
        for (int s = 0; s < S_; ++s) {
            const v4f ev = *(const v4f*)(eb + (size_t)s * ENC_);
            acc += ev * attn[s];
        }
        __bf16* dst = A2 + (size_t)b * KCAT_ + tid * 4;
        dst[0] = (__bf16)acc[0];
        dst[1] = (__bf16)acc[1];
        dst[2] = (__bf16)acc[2];
        dst[3] = (__bf16)acc[3];
    }
    // --- x_t copy into A2[b, 1024:1280]
    A2[(size_t)b * KCAT_ + ENC_ + tid] = xbf[((size_t)b * T_ + t) * E_ + tid];
}

// ---------------------------------------------------------------------------
// LSTM pointwise (gate order i,f,g,o). Updates c (f32), h (f32 + bf16 copies).
// ---------------------------------------------------------------------------
__global__ void lstm_cell_kernel(const float* __restrict__ z,
                                 float* __restrict__ c,
                                 float* __restrict__ hf,
                                 __bf16* __restrict__ hbf,
                                 __bf16* __restrict__ A2) {
    const int idx = blockIdx.x * blockDim.x + threadIdx.x;   // B*RNN = 65536
    const int b = idx >> 10, r = idx & 1023;
    const float* zb = z + (size_t)b * NGATE_;
    const float i = fast_sigmoid(zb[r]);
    const float f = fast_sigmoid(zb[RNN_ + r]);
    const float g = fast_tanh(zb[2 * RNN_ + r]);
    const float o = fast_sigmoid(zb[3 * RNN_ + r]);
    const float cn = f * c[idx] + i * g;
    const float hn = o * fast_tanh(cn);
    c[idx] = cn;
    hf[idx] = hn;
    const __bf16 hb = (__bf16)hn;
    hbf[idx] = hb;
    A2[(size_t)b * KCAT_ + ENC_ + E_ + r] = hb;
}

// ---------------------------------------------------------------------------
// One-time prep kernels
// ---------------------------------------------------------------------------
__global__ void transpose_cvt_kernel(const float* __restrict__ src,  // [K,N]
                                     __bf16* __restrict__ dst,       // [N,dstStride]
                                     int K, int N, int dstStride) {
    const size_t idx = (size_t)blockIdx.x * 256 + threadIdx.x;
    if (idx >= (size_t)K * N) return;
    const int k = (int)(idx / N);
    const int n = (int)(idx % N);
    dst[(size_t)n * dstStride + k] = (__bf16)src[idx];
}

__global__ void cvt_f32_bf16_kernel(const float* __restrict__ src,
                                    __bf16* __restrict__ dst, int n) {
    const int idx = blockIdx.x * 256 + threadIdx.x;
    if (idx < n) dst[idx] = (__bf16)src[idx];
}

__global__ void init_state_kernel(const float* __restrict__ hidden,
                                  const float* __restrict__ cell,
                                  __bf16* __restrict__ hbf,
                                  float* __restrict__ hf,
                                  float* __restrict__ c,
                                  __bf16* __restrict__ A2) {
    const int idx = blockIdx.x * 256 + threadIdx.x;   // 65536
    const int b = idx >> 10, r = idx & 1023;
    const float h = hidden[idx];
    hf[idx] = h;
    c[idx] = cell[idx];
    const __bf16 hb = (__bf16)h;
    hbf[idx] = hb;
    A2[(size_t)b * KCAT_ + ENC_ + E_ + r] = hb;
}

__global__ void copy_f32_kernel(const float* __restrict__ src,
                                float* __restrict__ dst, int n) {
    const int idx = blockIdx.x * 256 + threadIdx.x;
    if (idx < n) dst[idx] = src[idx];
}

// ---------------------------------------------------------------------------
extern "C" void kernel_launch(void* const* d_in, const int* in_sizes, int n_in,
                              void* d_out, int out_size, void* d_ws, size_t ws_size,
                              hipStream_t stream) {
    (void)in_sizes; (void)n_in; (void)out_size; (void)ws_size;
    const float* initial_input = (const float*)d_in[0];
    const float* hidden        = (const float*)d_in[1];
    const float* cell          = (const float*)d_in[2];
    const float* enc           = (const float*)d_in[3];
    const float* W1   = (const float*)d_in[4];
    const float* b1   = (const float*)d_in[5];
    const float* W2   = (const float*)d_in[6];
    const float* b2   = (const float*)d_in[7];
    const float* V    = (const float*)d_in[8];
    const float* bV   = (const float*)d_in[9];
    const float* Wker = (const float*)d_in[10];
    const float* Wrk  = (const float*)d_in[11];
    const float* bias = (const float*)d_in[12];
    const float* Wd   = (const float*)d_in[13];
    const float* bd   = (const float*)d_in[14];
    float* out = (float*)d_out;

    // ---- workspace carve (fully rewritten every call; ~61 MB) ----
    char* ws = (char*)d_ws;
    size_t off = 0;
    auto carve = [&](size_t bytes) -> void* {
        void* p = ws + off;
        off += (bytes + 255) & ~(size_t)255;
        return p;
    };
    __bf16* W1T   = (__bf16*)carve((size_t)ATT_ * RNN_ * 2);          // [ATT, RNN]
    __bf16* W2T   = (__bf16*)carve((size_t)ATT_ * ENC_ * 2);          // [ATT, ENC]
    __bf16* WcatT = (__bf16*)carve((size_t)NGATE_ * KCAT_ * 2);       // [4096, 2304]
    __bf16* WdT   = (__bf16*)carve((size_t)DENSE_ * RNN_ * 2);        // [8000, 1024]
    __bf16* xbf   = (__bf16*)carve((size_t)B_ * T_ * E_ * 2);
    __bf16* keysb = (__bf16*)carve((size_t)B_ * S_ * ATT_ * 2);
    float*  qbuf  = (float*)carve((size_t)B_ * ATT_ * 4);
    __bf16* A2    = (__bf16*)carve((size_t)B_ * KCAT_ * 2);           // [ctx|x_t|h]
    float*  zbuf  = (float*)carve((size_t)B_ * NGATE_ * 4);
    __bf16* hbf   = (__bf16*)carve((size_t)B_ * RNN_ * 2);
    float*  hf32  = (float*)carve((size_t)B_ * RNN_ * 4);
    float*  cbuf  = (float*)carve((size_t)B_ * RNN_ * 4);

    auto gblocks = [](int tilesM, int tilesN4) { return (tilesM * tilesN4 + 7) / 8; };

    // ---- phase 0: weight conversion / transposition to bf16 [N,K] ----
    transpose_cvt_kernel<<<(RNN_ * ATT_ + 255) / 256, 256, 0, stream>>>(W1, W1T, RNN_, ATT_, RNN_);
    transpose_cvt_kernel<<<(ENC_ * ATT_ + 255) / 256, 256, 0, stream>>>(W2, W2T, ENC_, ATT_, ENC_);
    transpose_cvt_kernel<<<((ENC_ + E_) * NGATE_ + 255) / 256, 256, 0, stream>>>(
        Wker, WcatT, ENC_ + E_, NGATE_, KCAT_);
    transpose_cvt_kernel<<<(RNN_ * NGATE_ + 255) / 256, 256, 0, stream>>>(
        Wrk, WcatT + (ENC_ + E_), RNN_, NGATE_, KCAT_);
    transpose_cvt_kernel<<<(RNN_ * DENSE_ + 255) / 256, 256, 0, stream>>>(Wd, WdT, RNN_, DENSE_, RNN_);
    cvt_f32_bf16_kernel<<<(B_ * T_ * E_ + 255) / 256, 256, 0, stream>>>(
        initial_input, xbf, B_ * T_ * E_);
    init_state_kernel<<<(B_ * RNN_) / 256, 256, 0, stream>>>(hidden, cell, hbf, hf32, cbuf, A2);

    // keys = enc @ W2 + b2   (M=8192, N=1024, K=1024; A fp32 converted in-flight)
    gemm_wmma_kernel<true, true><<<gblocks(512, 16), 256, 0, stream>>>(
        (const void*)enc, W2T, b2, (void*)keysb, B_ * S_, ATT_, ENC_, ENC_, ATT_);

    // ---- sequential decode ----
    for (int t = 0; t < T_; ++t) {
        // q = h @ W1 + b1
        gemm_wmma_kernel<false, false><<<gblocks(4, 16), 256, 0, stream>>>(
            (const void*)hbf, W1T, b1, (void*)qbuf, B_, ATT_, RNN_, RNN_, ATT_);
        // attention -> ctx, x_t into A2
        attn_ctx_kernel<<<B_, 256, 0, stream>>>(qbuf, keysb, V, bV, enc, xbf, A2, t);
        // z = [ctx|x_t|h] @ Wcat + bias   (K=2304, N=4096)
        gemm_wmma_kernel<false, false><<<gblocks(4, 64), 256, 0, stream>>>(
            (const void*)A2, WcatT, bias, (void*)zbuf, B_, NGATE_, KCAT_, KCAT_, NGATE_);
        // gates -> c,h
        lstm_cell_kernel<<<(B_ * RNN_) / 256, 256, 0, stream>>>(zbuf, cbuf, hf32, hbf, A2);
        // logits = h_new @ Wd + bd, stored directly at out[b, t, :] (ldc = T*DENSE)
        gemm_wmma_kernel<false, false><<<gblocks(4, 125), 256, 0, stream>>>(
            (const void*)hbf, WdT, bd, (void*)(out + (size_t)t * DENSE_),
            B_, DENSE_, RNN_, RNN_, (int)(T_ * DENSE_));
    }

    // ---- final h, c outputs ----
    float* out_h = out + (size_t)B_ * T_ * DENSE_;
    float* out_c = out_h + (size_t)B_ * RNN_;
    copy_f32_kernel<<<(B_ * RNN_ + 255) / 256, 256, 0, stream>>>(hf32, out_h, B_ * RNN_);
    copy_f32_kernel<<<(B_ * RNN_ + 255) / 256, 256, 0, stream>>>(cbuf, out_c, B_ * RNN_);
}
